// LinguisticDecoderLayer_29592324669675
// MI455X (gfx1250) — matrix operand order repaired and checked by reading, not compiled, not run on hardware
//
#include <hip/hip_runtime.h>
#include <hip/hip_bf16.h>
#include <math.h>

typedef __attribute__((ext_vector_type(16))) __bf16 bf16x16;
typedef __attribute__((ext_vector_type(8)))  __bf16 bf16x8;
typedef __attribute__((ext_vector_type(8)))  float  f32x8;

#define WMMA_BF16 __builtin_amdgcn_wmma_f32_16x16x32_bf16

__device__ __forceinline__ __bf16 f2bf(float f) {
  unsigned u = __builtin_bit_cast(unsigned, f);
  u += 0x7FFFu + ((u >> 16) & 1u);              // round-to-nearest-even
  unsigned short s = (unsigned short)(u >> 16);
  return __builtin_bit_cast(__bf16, s);
}

__device__ __forceinline__ f32x8 zero8() {
  f32x8 z;
#pragma unroll
  for (int i = 0; i < 8; ++i) z[i] = 0.0f;
  return z;
}

// A-fragment (16x32 bf16): per ISA, elements 0..7 -> K = half*8 + 0..7,
// elements 8..15 -> K = half*8 + 16..23 (two 16B LDS loads per lane).
__device__ __forceinline__ bf16x16 ldfragA(const __bf16* rowBase, int hh) {
  const bf16x8* p = (const bf16x8*)(rowBase + hh * 8);
  bf16x8 lo = p[0];
  bf16x8 hi = p[2];   // +16 elements
  return __builtin_shufflevector(lo, hi, 0,1,2,3,4,5,6,7,8,9,10,11,12,13,14,15);
}

// B-fragment (32x16 bf16): elements 0..7 -> K = half*8 + 0..7,
// elements 8..15 -> K = half*8 + 16..23 (same order as A; two 16B loads from [n][k]).
__device__ __forceinline__ bf16x16 ldfragB(const __bf16* rowBase, int hh) {
  const bf16x8* p = (const bf16x8*)(rowBase + hh * 8);
  bf16x8 lo = p[0];
  bf16x8 hi = p[2];
  return __builtin_shufflevector(lo, hi, 0,1,2,3,4,5,6,7,8,9,10,11,12,13,14,15);
}

// CDNA5 async copy: 16 bytes per lane, global -> LDS, tracked by ASYNCcnt.
__device__ __forceinline__ void async_cp16(const void* gaddr, void* ldsPtr) {
  asm volatile("global_load_async_to_lds_b128 %0, %1, off"
               :
               : "v"((unsigned)(uintptr_t)ldsPtr), "v"(gaddr)
               : "memory");
}
__device__ __forceinline__ void wait_async() {
  asm volatile("s_wait_asynccnt 0x0" ::: "memory");
}

// ---------------------------------------------------------------------------
// Weight prep: fp32 W[K x N] -> bf16 Wt[N x K] (transpose, one time).
// ---------------------------------------------------------------------------
__global__ __launch_bounds__(256) void convT_kernel(
    const float* __restrict__ W, __bf16* __restrict__ Wt, int K, int N)
{
  const int idx = blockIdx.x * 256 + threadIdx.x;
  const int k = idx / N;
  const int n = idx - k * N;
  Wt[(size_t)n * K + k] = f2bf(W[idx]);
}

__global__ __launch_bounds__(256) void conv_kernel(
    const float* __restrict__ X, __bf16* __restrict__ Y)
{
  const int idx = blockIdx.x * 256 + threadIdx.x;
  Y[idx] = f2bf(X[idx]);
}

// ---------------------------------------------------------------------------
// LayerNorm: one 1024-wide row per wave32, fp32 in, bf16 out.
// ---------------------------------------------------------------------------
__global__ __launch_bounds__(256) void layernorm_kernel(
    const float* __restrict__ x, const float* __restrict__ g,
    const float* __restrict__ be, __bf16* __restrict__ y)
{
  const int lane = threadIdx.x & 31;
  const int wid  = threadIdx.x >> 5;
  const int row  = blockIdx.x * 8 + wid;
  const float* xr = x + (size_t)row * 1024;
  float v[32];
  float s = 0.0f;
#pragma unroll
  for (int i = 0; i < 32; ++i) { v[i] = xr[lane + i * 32]; s += v[i]; }
#pragma unroll
  for (int off = 1; off < 32; off <<= 1) s += __shfl_xor(s, off, 32);
  const float mu = s * (1.0f / 1024.0f);
  float q = 0.0f;
#pragma unroll
  for (int i = 0; i < 32; ++i) { float d = v[i] - mu; q += d * d; }
#pragma unroll
  for (int off = 1; off < 32; off <<= 1) q += __shfl_xor(q, off, 32);
  const float inv = rsqrtf(q * (1.0f / 1024.0f) + 1e-5f);
  __bf16* yr = y + (size_t)row * 1024;
#pragma unroll
  for (int i = 0; i < 32; ++i) {
    int c = lane + i * 32;
    yr[c] = f2bf((v[i] - mu) * inv * g[c] + be[c]);
  }
}

// ---------------------------------------------------------------------------
// GEMM: act(A[MxK]bf16 * Wt[NxK]bf16^T + bias) (+resid). 128x128 block tile,
// 8 waves (4x2), each 32x64 -> 8 WMMA per k-step. Double-buffered LDS filled
// by global_load_async_to_lds_b128.
// ---------------------------------------------------------------------------
__global__ __launch_bounds__(256) void gemm_kernel(
    const __bf16* __restrict__ A,   // M x K
    const __bf16* __restrict__ Wt,  // N x K (pre-transposed)
    const float* __restrict__ bias,
    const float* __restrict__ resid,
    float* __restrict__ Cf,         // fp32 out (nullable)
    __bf16* __restrict__ Cb,        // bf16 out (nullable)
    int K, int N, int act)
{
  __shared__ __bf16 As[2][128 * 40];  // [m][k], row stride 40 (80B, 16B aligned)
  __shared__ __bf16 Bs[2][128 * 40];  // [n][k]
  const int tid  = threadIdx.x;
  const int lane = tid & 31;
  const int wid  = tid >> 5;
  const int hh   = lane >> 4;
  const int l16  = lane & 15;
  const int wm   = wid & 3;           // 4 row-blocks of 32
  const int wn   = wid >> 2;          // 2 col-blocks of 64
  const int rowBase = blockIdx.y * 128;
  const int colBase = blockIdx.x * 128;

  f32x8 acc[2][4];
#pragma unroll
  for (int a = 0; a < 2; ++a)
#pragma unroll
    for (int t = 0; t < 4; ++t) acc[a][t] = zero8();

  // issue async fill of one 128x32 A tile + 128x32 B tile (4 x 16B per thread)
  auto issueTile = [&](int buf, int k0) {
#pragma unroll
    for (int t = 0; t < 2; ++t) {
      int c = tid + t * 256;
      int row = c >> 2, sub = c & 3;
      async_cp16(A + (size_t)(rowBase + row) * K + k0 + sub * 8,
                 &As[buf][row * 40 + sub * 8]);
    }
#pragma unroll
    for (int t = 0; t < 2; ++t) {
      int c = tid + t * 256;
      int row = c >> 2, sub = c & 3;
      async_cp16(Wt + (size_t)(colBase + row) * K + k0 + sub * 8,
                 &Bs[buf][row * 40 + sub * 8]);
    }
  };

  issueTile(0, 0);
  int buf = 0;
  for (int k0 = 0; k0 < K; k0 += 32) {
    wait_async();
    __syncthreads();
    if (k0 + 32 < K) issueTile(buf ^ 1, k0 + 32);
    bf16x16 af[2], bfr[4];
#pragma unroll
    for (int a = 0; a < 2; ++a)
      af[a] = ldfragA(&As[buf][(wm * 32 + a * 16 + l16) * 40], hh);
#pragma unroll
    for (int t = 0; t < 4; ++t)
      bfr[t] = ldfragB(&Bs[buf][(wn * 64 + t * 16 + l16) * 40], hh);
#pragma unroll
    for (int a = 0; a < 2; ++a)
#pragma unroll
      for (int t = 0; t < 4; ++t)
        acc[a][t] = WMMA_BF16(false, af[a], false, bfr[t], (short)0, acc[a][t], false, false);
    buf ^= 1;
  }

#pragma unroll
  for (int a = 0; a < 2; ++a) {
#pragma unroll
    for (int r = 0; r < 8; ++r) {
      const int row = rowBase + wm * 32 + a * 16 + r + hh * 8;
#pragma unroll
      for (int t = 0; t < 4; ++t) {
        const int col = colBase + wn * 64 + t * 16 + l16;
        float v = acc[a][t][r] + bias[col];
        if (act == 1) v = 0.5f * v * (1.0f + erff(v * 0.70710678118654752f));
        if (resid) v += resid[(size_t)row * N + col];
        if (Cf) Cf[(size_t)row * N + col] = v;
        if (Cb) Cb[(size_t)row * N + col] = f2bf(v);
      }
    }
  }
}

// ---------------------------------------------------------------------------
// RoPE in place on bf16 q,k slices of fused QKV (B,S,3*1024).
// ---------------------------------------------------------------------------
__global__ __launch_bounds__(256) void rope_kernel(
    __bf16* __restrict__ qkv, const float* __restrict__ cosT,
    const float* __restrict__ sinT)
{
  const int idx = blockIdx.x * 256 + threadIdx.x;
  const int i = idx & 31;            // freq index (HD/2 = 32)
  const int h = (idx >> 5) & 15;
  const int w = (idx >> 9) & 1;      // 0 = q, 1 = k
  const int s = (idx >> 10) & 2047;
  const int b = idx >> 21;
  __bf16* p = qkv + ((size_t)(b * 2048 + s)) * 3072 + w * 1024 + h * 64 + 2 * i;
  const float c  = cosT[s * 32 + i];
  const float sn = sinT[s * 32 + i];
  const float xe = (float)p[0];
  const float xo = (float)p[1];
  p[0] = f2bf(xe * c - xo * sn);
  p[1] = f2bf(xe * sn + xo * c);
}

// ---------------------------------------------------------------------------
// Flash attention (causal or full), bf16 q/k/v, bf16 out, head dim 64.
// 128 q-rows per block (8 waves x 16 rows), online softmax, WMMA everywhere.
// ---------------------------------------------------------------------------
__global__ __launch_bounds__(256) void attn_kernel(
    const __bf16* __restrict__ Qp, const __bf16* __restrict__ Kp,
    const __bf16* __restrict__ Vp, __bf16* __restrict__ Op,
    int qStride, int kStride, int Sq, int Sk, int causal)
{
  __shared__ __bf16 Ks[64 * 72];        // [key][d]
  __shared__ __bf16 Vs[64 * 72];        // [d][key] (transposed)
  __shared__ __bf16 Ps[8 * 16 * 72];    // per-wave P scratch [qrow][key]
  const int tid  = threadIdx.x;
  const int lane = tid & 31;
  const int wid  = tid >> 5;
  const int hh   = lane >> 4;
  const int l16  = lane & 15;
  const int b = blockIdx.y >> 4;
  const int h = blockIdx.y & 15;
  const int qbase = blockIdx.x * 128;
  const float scale = 0.125f;           // 1/sqrt(64)

  const __bf16* qptr = Qp + (size_t)b * Sq * qStride + h * 64;
  const __bf16* kptr = Kp + (size_t)b * Sk * kStride + h * 64;
  const __bf16* vptr = Vp + (size_t)b * Sk * kStride + h * 64;

  // Q A-fragments (16x64 per wave -> 2 k-steps), gathered per ISA A layout.
  const int qrow = qbase + wid * 16 + l16;
  bf16x16 qf[2];
#pragma unroll
  for (int ks = 0; ks < 2; ++ks) {
#pragma unroll
    for (int e = 0; e < 16; ++e) {
      int kk = ks * 32 + ((e & 8) ? (e + 8) : e) + hh * 8;
      qf[ks][e] = qptr[(size_t)qrow * qStride + kk];
    }
  }

  f32x8 o[4];
  float mrow[8], lrow[8];
#pragma unroll
  for (int dt = 0; dt < 4; ++dt) o[dt] = zero8();
#pragma unroll
  for (int r = 0; r < 8; ++r) { mrow[r] = -3.0e38f; lrow[r] = 0.0f; }

  const int jend = causal ? (qbase + 128) : Sk;
  for (int jb = 0; jb < jend; jb += 64) {
    // K rows: vectorized 16B copies; V: transposed elementwise.
    for (int i = tid; i < 64 * 8; i += 256) {
      int kr = i >> 3, sub = i & 7;
      *(bf16x8*)&Ks[kr * 72 + sub * 8] =
          *(const bf16x8*)&kptr[(size_t)(jb + kr) * kStride + sub * 8];
    }
    for (int i = tid; i < 64 * 64; i += 256) {
      int kr = i >> 6, c = i & 63;
      Vs[c * 72 + kr] = vptr[(size_t)(jb + kr) * kStride + c];
    }
    __syncthreads();

    // S = Q K^T : 4 n-tiles x 2 k-steps
    f32x8 s[4];
#pragma unroll
    for (int nt = 0; nt < 4; ++nt) s[nt] = zero8();
#pragma unroll
    for (int ks = 0; ks < 2; ++ks) {
#pragma unroll
      for (int nt = 0; nt < 4; ++nt) {
        bf16x16 bfrag = ldfragB(&Ks[(nt * 16 + l16) * 72 + ks * 32], hh);
        s[nt] = WMMA_BF16(false, qf[ks], false, bfrag, (short)0, s[nt], false, false);
      }
    }

    // Online softmax. Row r lives in the 16 lanes of one half-wave.
#pragma unroll
    for (int r = 0; r < 8; ++r) {
      const int row = qbase + wid * 16 + r + hh * 8;
      float sc[4];
      float mx = mrow[r];
#pragma unroll
      for (int nt = 0; nt < 4; ++nt) {
        int col = jb + nt * 16 + l16;
        float v = s[nt][r] * scale;
        if (causal && (col > row)) v = -1.0e9f;
        sc[nt] = v;
        mx = fmaxf(mx, v);
      }
#pragma unroll
      for (int off = 1; off < 16; off <<= 1) mx = fmaxf(mx, __shfl_xor(mx, off, 32));
      const float corr = __expf(mrow[r] - mx);
      float rs = 0.0f;
#pragma unroll
      for (int nt = 0; nt < 4; ++nt) { sc[nt] = __expf(sc[nt] - mx); rs += sc[nt]; }
#pragma unroll
      for (int off = 1; off < 16; off <<= 1) rs += __shfl_xor(rs, off, 32);
      mrow[r] = mx;
      lrow[r] = lrow[r] * corr + rs;
#pragma unroll
      for (int dt = 0; dt < 4; ++dt) o[dt][r] *= corr;
#pragma unroll
      for (int nt = 0; nt < 4; ++nt)
        Ps[(wid * 16 + r + hh * 8) * 72 + nt * 16 + l16] = f2bf(sc[nt]);
    }

    // O += P V : P re-read in A layout from this wave's private LDS region.
#pragma unroll
    for (int ks = 0; ks < 2; ++ks) {
      bf16x16 pfrag = ldfragA(&Ps[(wid * 16 + l16) * 72 + ks * 32], hh);
#pragma unroll
      for (int dt = 0; dt < 4; ++dt) {
        bf16x16 vfrag = ldfragB(&Vs[(dt * 16 + l16) * 72 + ks * 32], hh);
        o[dt] = WMMA_BF16(false, pfrag, false, vfrag, (short)0, o[dt], false, false);
      }
    }
    __syncthreads();
  }

#pragma unroll
  for (int r = 0; r < 8; ++r) {
    const int row = qbase + wid * 16 + r + hh * 8;
    const float inv = 1.0f / lrow[r];
#pragma unroll
    for (int dt = 0; dt < 4; ++dt) {
      Op[(size_t)(b * Sq + row) * 1024 + h * 64 + dt * 16 + l16] =
          f2bf(o[dt][r] * inv);
    }
  }
}

// ---------------------------------------------------------------------------
extern "C" void kernel_launch(void* const* d_in, const int* in_sizes, int n_in,
                              void* d_out, int out_size, void* d_ws, size_t ws_size,
                              hipStream_t stream) {
  (void)in_sizes; (void)n_in; (void)out_size; (void)ws_size;
  const float* tgt   = (const float*)d_in[0];
  const float* mem   = (const float*)d_in[1];
  const float* rcos  = (const float*)d_in[2];
  const float* rsin  = (const float*)d_in[3];
  const float* W_qkv = (const float*)d_in[4];
  const float* b_qkv = (const float*)d_in[5];
  const float* W_o   = (const float*)d_in[6];
  const float* b_o   = (const float*)d_in[7];
  const float* Wq_c  = (const float*)d_in[8];
  const float* bq_c  = (const float*)d_in[9];
  const float* Wk_c  = (const float*)d_in[10];
  const float* bk_c  = (const float*)d_in[11];
  const float* Wv_c  = (const float*)d_in[12];
  const float* bv_c  = (const float*)d_in[13];
  const float* W_co  = (const float*)d_in[14];
  const float* b_co  = (const float*)d_in[15];
  const float* W1    = (const float*)d_in[16];
  const float* b1    = (const float*)d_in[17];
  const float* W2    = (const float*)d_in[18];
  const float* b2    = (const float*)d_in[19];
  const float* g1    = (const float*)d_in[20];
  const float* be1   = (const float*)d_in[21];
  const float* g2    = (const float*)d_in[22];
  const float* be2   = (const float*)d_in[23];
  const float* g3    = (const float*)d_in[24];
  const float* be3   = (const float*)d_in[25];
  float* out = (float*)d_out;

  const int BS = 2 * 2048;

  // workspace layout (all counts multiples of 128 -> alignment preserved)
  char* base = (char*)d_ws;
  auto allocB = [&](size_t elems) { __bf16* p = (__bf16*)base; base += elems * 2; return p; };
  __bf16* Wt_qkv = allocB((size_t)3072 * 1024);
  __bf16* Wt_o   = allocB((size_t)1024 * 1024);
  __bf16* Wt_qc  = allocB((size_t)1024 * 1024);
  __bf16* Wt_kc  = allocB((size_t)1024 * 1024);
  __bf16* Wt_vc  = allocB((size_t)1024 * 1024);
  __bf16* Wt_co  = allocB((size_t)1024 * 1024);
  __bf16* W1t    = allocB((size_t)4096 * 1024);
  __bf16* W2t    = allocB((size_t)1024 * 4096);
  __bf16* mem_bf = allocB((size_t)128 * 1024);
  __bf16* x_bf   = allocB((size_t)BS * 1024);
  __bf16* qkv_bf = allocB((size_t)BS * 3072);
  __bf16* sa_bf  = allocB((size_t)BS * 1024);
  __bf16* qc_bf  = allocB((size_t)BS * 1024);
  __bf16* kc_bf  = allocB((size_t)128 * 1024);
  __bf16* vc_bf  = allocB((size_t)128 * 1024);
  __bf16* h1_bf  = allocB((size_t)BS * 4096);
  float* t1 = (float*)base; base += (size_t)BS * 1024 * 4;
  float* t2 = (float*)base; base += (size_t)BS * 1024 * 4;

  dim3 blk(256);

  // one-time weight conversion + transpose to bf16 [N x K]
  convT_kernel<<<(1024 * 3072) / 256, blk, 0, stream>>>(W_qkv, Wt_qkv, 1024, 3072);
  convT_kernel<<<(1024 * 1024) / 256, blk, 0, stream>>>(W_o,  Wt_o,  1024, 1024);
  convT_kernel<<<(1024 * 1024) / 256, blk, 0, stream>>>(Wq_c, Wt_qc, 1024, 1024);
  convT_kernel<<<(1024 * 1024) / 256, blk, 0, stream>>>(Wk_c, Wt_kc, 1024, 1024);
  convT_kernel<<<(1024 * 1024) / 256, blk, 0, stream>>>(Wv_c, Wt_vc, 1024, 1024);
  convT_kernel<<<(1024 * 1024) / 256, blk, 0, stream>>>(W_co, Wt_co, 1024, 1024);
  convT_kernel<<<(1024 * 4096) / 256, blk, 0, stream>>>(W1, W1t, 1024, 4096);
  convT_kernel<<<(4096 * 1024) / 256, blk, 0, stream>>>(W2, W2t, 4096, 1024);
  conv_kernel<<<(128 * 1024) / 256, blk, 0, stream>>>(mem, mem_bf);

  // x = LN(tgt); qkv = x @ W_qkv + b ; RoPE ; causal self-attention
  layernorm_kernel<<<BS / 8, blk, 0, stream>>>(tgt, g1, be1, x_bf);
  gemm_kernel<<<dim3(3072 / 128, BS / 128), blk, 0, stream>>>(
      x_bf, Wt_qkv, b_qkv, nullptr, nullptr, qkv_bf, 1024, 3072, 0);
  rope_kernel<<<(BS * 1024) / 256, blk, 0, stream>>>(qkv_bf, rcos, rsin);
  attn_kernel<<<dim3(2048 / 128, 32), blk, 0, stream>>>(
      qkv_bf, qkv_bf + 1024, qkv_bf + 2048, sa_bf, 3072, 3072, 2048, 2048, 1);
  // t1 = tgt + sa @ W_o
  gemm_kernel<<<dim3(1024 / 128, BS / 128), blk, 0, stream>>>(
      sa_bf, Wt_o, b_o, tgt, t1, nullptr, 1024, 1024, 0);

  // cross attention
  layernorm_kernel<<<BS / 8, blk, 0, stream>>>(t1, g2, be2, x_bf);
  gemm_kernel<<<dim3(1024 / 128, BS / 128), blk, 0, stream>>>(
      x_bf, Wt_qc, bq_c, nullptr, nullptr, qc_bf, 1024, 1024, 0);
  gemm_kernel<<<dim3(1024 / 128, 1), blk, 0, stream>>>(
      mem_bf, Wt_kc, bk_c, nullptr, nullptr, kc_bf, 1024, 1024, 0);
  gemm_kernel<<<dim3(1024 / 128, 1), blk, 0, stream>>>(
      mem_bf, Wt_vc, bv_c, nullptr, nullptr, vc_bf, 1024, 1024, 0);
  attn_kernel<<<dim3(2048 / 128, 32), blk, 0, stream>>>(
      qc_bf, kc_bf, vc_bf, sa_bf, 1024, 1024, 2048, 64, 0);
  // t2 = t1 + co @ W_co
  gemm_kernel<<<dim3(1024 / 128, BS / 128), blk, 0, stream>>>(
      sa_bf, Wt_co, b_co, t1, t2, nullptr, 1024, 1024, 0);

  // FFN
  layernorm_kernel<<<BS / 8, blk, 0, stream>>>(t2, g3, be3, x_bf);
  gemm_kernel<<<dim3(4096 / 128, BS / 128), blk, 0, stream>>>(
      x_bf, W1t, b1, nullptr, nullptr, h1_bf, 1024, 4096, 1);
  gemm_kernel<<<dim3(1024 / 128, BS / 128), blk, 0, stream>>>(
      h1_bf, W2t, b2, t2, out, nullptr, 4096, 1024, 0);
}
